// ToMEBlock_56470230008224
// MI455X (gfx1250) — matrix operand order repaired and checked by reading, not compiled, not run on hardware
//
#include <hip/hip_runtime.h>
#include <hip/hip_bf16.h>

// ---------------------------------------------------------------------------
// ToMe token-merge block for MI455X (gfx1250, wave32).
// Shapes fixed by the reference: B=8, T=2048, C=1024, H=64, Ta=Tb=1024, r=512.
// ---------------------------------------------------------------------------

typedef __attribute__((ext_vector_type(16))) _Float16 v16h;
typedef __attribute__((ext_vector_type(8)))  _Float16 v8h;
typedef __attribute__((ext_vector_type(8)))  float    v8f;
typedef __attribute__((ext_vector_type(4)))  float    v4f;

constexpr int B_  = 8;
constexpr int T_  = 2048;
constexpr int C_  = 1024;
constexpr int H_  = 64;
constexpr int Ta_ = 1024;   // even tokens per batch
constexpr int Tb_ = 1024;   // odd  tokens per batch

// ---------------------------------------------------------------------------
// K1: m = metric / ||metric||, stored as f16 row-major (B*T rows of 64 halves).
// One wave per row; 8 waves (256 threads) per block.
// ---------------------------------------------------------------------------
__global__ __launch_bounds__(256) void tome_normalize_kernel(
    const float* __restrict__ metric, _Float16* __restrict__ m16) {
  const int wave = threadIdx.x >> 5;
  const int lane = threadIdx.x & 31;
  const int row  = blockIdx.x * 8 + wave;          // 0 .. B*T-1
  const float* src = metric + (size_t)row * H_;
  float v0 = src[lane];
  float v1 = src[lane + 32];
  float ss = v0 * v0 + v1 * v1;
#pragma unroll
  for (int m = 16; m >= 1; m >>= 1) ss += __shfl_xor(ss, m, 32);
  const float inv = rsqrtf(ss);
  _Float16* dst = m16 + (size_t)row * H_;
  dst[lane]      = (_Float16)(v0 * inv);
  dst[lane + 32] = (_Float16)(v1 * inv);
}

// ---------------------------------------------------------------------------
// K2: scores = a . b^T with fused row max/argmax.
// One wave per (batch, 16-row tile of a). For each of 64 j-tiles:
//   D(16x16) = A(16x64) * B(64x16) via two v_wmma_f32_16x16x32_f16,
// spill the f32 tile to LDS, then ALL 32 lanes reduce: lane l scans row
// (l&15), cols hi*8..hi*8+7 (two ds_load_b128), and the two half-row
// candidates are merged at the end with one shfl_xor(16) using the
// (value desc, index asc) tie-break == exact first-argmax semantics.
//
// A fragment layout (ISA 7.12.2, 16-bit A 16x32): lane l holds M=l&15;
//   half-wave 0: K = k0+{0..7}, k0+16+{0..7}; half-wave 1: +8.
// B fragment layout (same K order as A): lane l holds N=l&15;
//   half-wave 0: K = k0+{0..7}, k0+16+{0..7}; half-wave 1: +8.
// D layout: VGPR r, lanes 0..15 -> M=r, N=lane; lanes 16..31 -> M=8+r.
// ---------------------------------------------------------------------------
__global__ __launch_bounds__(32) void tome_match_kernel(
    const _Float16* __restrict__ m16,
    float* __restrict__ node_max, int* __restrict__ node_idx) {
  __shared__ float sc[256];

  const int wg   = blockIdx.x;          // 0 .. B*(Ta/16)-1
  const int bb   = wg >> 6;             // batch
  const int i0   = (wg & 63) << 4;      // a-row tile base
  const int lane = threadIdx.x;
  const int lm   = lane & 15;
  const int hi   = lane >> 4;

  // A fragments for rows i0..i0+15 (even tokens t = 2*i), K = 0..63.
  const _Float16* arow = m16 + ((size_t)(bb * T_ + 2 * (i0 + lm)) * H_);
  union { v16h v; v8h h[2]; } ua;
  ua.h[0] = *(const v8h*)(arow + 0  + hi * 8);
  ua.h[1] = *(const v8h*)(arow + 16 + hi * 8);
  const v16h a0 = ua.v;
  ua.h[0] = *(const v8h*)(arow + 32 + hi * 8);
  ua.h[1] = *(const v8h*)(arow + 48 + hi * 8);
  const v16h a1 = ua.v;

  float best  = -3.4e38f;
  int   bestj = 0x7fffffff;

  for (int jt = 0; jt < Tb_ / 16; ++jt) {
    // B fragment: columns j = jt*16 + N (odd tokens t = 2*j+1).
    const _Float16* brow = m16 + ((size_t)(bb * T_ + 2 * (jt * 16 + lm) + 1) * H_);
    union { v16h v; v8h h[2]; } ub;
    ub.h[0] = *(const v8h*)(brow + 0  + hi * 8);
    ub.h[1] = *(const v8h*)(brow + 16 + hi * 8);
    const v16h b0 = ub.v;
    ub.h[0] = *(const v8h*)(brow + 32 + hi * 8);
    ub.h[1] = *(const v8h*)(brow + 48 + hi * 8);
    const v16h b1 = ub.v;

    v8f c = {};
    c = __builtin_amdgcn_wmma_f32_16x16x32_f16(false, a0, false, b0,
                                               (short)0, c, false, false);
    c = __builtin_amdgcn_wmma_f32_16x16x32_f16(false, a1, false, b1,
                                               (short)0, c, false, false);

#pragma unroll
    for (int rr = 0; rr < 8; ++rr) sc[(rr + hi * 8) * 16 + lm] = c[rr];
    __syncthreads();

    // Lane l reduces row lm over cols hi*8 .. hi*8+7 (two b128 LDS loads).
    {
      const v4f* rowp = (const v4f*)&sc[lm * 16 + hi * 8];
      const v4f q0 = rowp[0];
      const v4f q1 = rowp[1];
      const int jb = jt * 16 + hi * 8;
      const float vv[8] = {q0.x, q0.y, q0.z, q0.w, q1.x, q1.y, q1.z, q1.w};
#pragma unroll
      for (int e = 0; e < 8; ++e) {
        if (vv[e] > best) { best = vv[e]; bestj = jb + e; }  // first argmax
      }
    }
    __syncthreads();
  }

  // Merge the two half-row candidates (cols 0-7 vs 8-15 of every tile).
  {
    const float ob = __shfl_xor(best, 16, 32);
    const int   oj = __shfl_xor(bestj, 16, 32);
    if (ob > best || (ob == best && oj < bestj)) { best = ob; bestj = oj; }
  }

  if (lane < 16) {
    const int row = bb * Ta_ + i0 + lane;        // flat (B*Ta) row
    node_max[row] = best;
    node_idx[row] = bestj;
  }
}

// ---------------------------------------------------------------------------
// K3: exact rank of each node_max under argsort(-node_max) semantics
// (descending value, ascending index on ties). rank < r*B => merged src.
// All 8192 values staged in LDS; scanned as v4f (ds_load_b128).
// ---------------------------------------------------------------------------
__global__ __launch_bounds__(256) void tome_rank_kernel(
    const float* __restrict__ node_max, const int* __restrict__ rp,
    int* __restrict__ is_src) {
  __shared__ float s_nm[B_ * Ta_];                // 32 KB
  const int tid = threadIdx.x;
  {
    const v4f* src = (const v4f*)node_max;
    v4f* dst = (v4f*)s_nm;
    for (int k = tid; k < (B_ * Ta_) / 4; k += 256) dst[k] = src[k];
  }
  __syncthreads();

  const int   t   = blockIdx.x * 256 + tid;
  const float val = s_nm[t];
  int rank = 0;
  const v4f* sv = (const v4f*)s_nm;
  for (int u4 = 0; u4 < (B_ * Ta_) / 4; ++u4) {
    const v4f o = sv[u4];
    const int u = u4 * 4;
    rank += (o.x > val) || (o.x == val && (u + 0) < t);
    rank += (o.y > val) || (o.y == val && (u + 1) < t);
    rank += (o.z > val) || (o.z == val && (u + 2) < t);
    rank += (o.w > val) || (o.w == val && (u + 3) < t);
  }
  const int rb = rp[0] * B_;
  is_src[t] = (rank < rb) ? 1 : 0;
}

// ---------------------------------------------------------------------------
// K4: per-batch stable compaction positions via inclusive scan of merged flag.
//   pos_mrg[i] = #merged before i, pos_unm[i] = #kept before i,
//   n_unm[b]   = kept even-token count.
// ---------------------------------------------------------------------------
__global__ __launch_bounds__(1024) void tome_scan_kernel(
    const int* __restrict__ is_src, int* __restrict__ pos_unm,
    int* __restrict__ pos_mrg, int* __restrict__ n_unm) {
  __shared__ int s[Ta_];
  const int b = blockIdx.x, i = threadIdx.x;
  const int m = is_src[b * Ta_ + i];
  s[i] = m;
  __syncthreads();
  for (int off = 1; off < Ta_; off <<= 1) {
    const int v = (i >= off) ? s[i - off] : 0;
    __syncthreads();
    s[i] += v;
    __syncthreads();
  }
  const int incl = s[i];
  pos_mrg[b * Ta_ + i] = incl - m;
  pos_unm[b * Ta_ + i] = i - (incl - m);
  if (i == Ta_ - 1) n_unm[b] = Ta_ - incl;
}

// ---------------------------------------------------------------------------
// K5: even (src-half) tokens. Kept rows -> out rows [0, n_unm);
// merged rows -> zeroed tail rows [n_unm+Tb, T), size=0, pad=1.
// 256 threads x 4 contiguous floats = one b128 load + one b128 store each.
// ---------------------------------------------------------------------------
__global__ __launch_bounds__(256) void tome_out_even_kernel(
    const float* __restrict__ x, const float* __restrict__ size,
    const int* __restrict__ is_src, const int* __restrict__ pos_unm,
    const int* __restrict__ pos_mrg, const int* __restrict__ n_unm,
    float* __restrict__ out) {
  const int bi = blockIdx.x;            // 0 .. B*Ta-1
  const int b  = bi >> 10;
  const int i  = bi & (Ta_ - 1);
  const int tsrc   = 2 * i;
  const int merged = is_src[bi];
  const int orow   = merged ? (n_unm[b] + Tb_ + pos_mrg[bi]) : pos_unm[bi];

  const float s     = size[b * T_ + tsrc];
  const float scale = merged ? 0.0f : s / (s + 1e-4f);   // (x*s)/(s+eps)

  const float* xin = x   + ((size_t)(b * T_ + tsrc)) * C_;
  float*       xo  = out + ((size_t)(b * T_) + orow) * C_;
  const int c = threadIdx.x * 4;                          // 256*4 == C
  const v4f xv = *(const v4f*)(xin + c);
  *(v4f*)(xo + c) = xv * scale;

  if (threadIdx.x == 0) {
    float* size_out = out + (size_t)B_ * T_ * C_;
    float* pad_out  = size_out + (size_t)B_ * T_;
    size_out[b * T_ + orow] = merged ? 0.0f : s;
    pad_out[b * T_ + orow]  = merged ? 1.0f : 0.0f;
  }
}

// ---------------------------------------------------------------------------
// K6: dst (odd-half) tokens -> out rows [n_unm, n_unm+Tb).
// Thread 0 builds the ordered list of merged srcs targeting this dst
// (deterministic summation order, no float atomics), then the block sums
// channels as v4f: (x_dst*s_dst + sum x_i*s_i) / (s_dst + sum s_i + 1e-4).
// ---------------------------------------------------------------------------
__global__ __launch_bounds__(256) void tome_out_dst_kernel(
    const float* __restrict__ x, const float* __restrict__ size,
    const int* __restrict__ is_src, const int* __restrict__ node_idx,
    const int* __restrict__ n_unm, float* __restrict__ out) {
  __shared__ int   list[Ta_];
  __shared__ float slist[Ta_];
  __shared__ int   cnt;
  __shared__ float ssum;

  const int bi = blockIdx.x;            // 0 .. B*Tb-1
  const int b  = bi >> 10;
  const int j  = bi & (Tb_ - 1);
  const int tdst = 2 * j + 1;

  if (threadIdx.x == 0) {
    int   n  = 0;
    float ss = size[b * T_ + tdst];
    for (int i = 0; i < Ta_; ++i) {
      const int idx = b * Ta_ + i;
      if (is_src[idx] && node_idx[idx] == j) {
        const float sv = size[b * T_ + 2 * i];
        list[n]  = i;
        slist[n] = sv;
        ss += sv;
        ++n;
      }
    }
    cnt  = n;
    ssum = ss;
  }
  __syncthreads();

  const int   n    = cnt;
  const float sden = 1.0f / (ssum + 1e-4f);
  const float sdst = size[b * T_ + tdst];
  const int   orow = n_unm[b] + j;

  const float* xdst = x   + ((size_t)(b * T_ + tdst)) * C_;
  float*       xo   = out + ((size_t)(b * T_) + orow) * C_;
  const int c = threadIdx.x * 4;                          // 256*4 == C
  v4f acc = *(const v4f*)(xdst + c) * sdst;
  for (int k = 0; k < n; ++k) {
    const v4f xs = *(const v4f*)(x + ((size_t)(b * T_ + 2 * list[k])) * C_ + c);
    acc += xs * slist[k];
  }
  *(v4f*)(xo + c) = acc * sden;

  if (threadIdx.x == 0) {
    float* size_out = out + (size_t)B_ * T_ * C_;
    float* pad_out  = size_out + (size_t)B_ * T_;
    size_out[b * T_ + orow] = ssum;
    pad_out[b * T_ + orow]  = 0.0f;
  }
}

// ---------------------------------------------------------------------------
// Host launcher. d_in: [x, metric, size, r]; d_out: [x_out | size_s | pad_s].
// Workspace: f16 normalized metric (2 MB) + small index/score arrays.
// ---------------------------------------------------------------------------
extern "C" void kernel_launch(void* const* d_in, const int* in_sizes, int n_in,
                              void* d_out, int out_size, void* d_ws, size_t ws_size,
                              hipStream_t stream) {
  (void)in_sizes; (void)n_in; (void)out_size; (void)ws_size;

  const float* x      = (const float*)d_in[0];
  const float* metric = (const float*)d_in[1];
  const float* size   = (const float*)d_in[2];
  const int*   rp     = (const int*)d_in[3];
  float*       out    = (float*)d_out;

  char* ws = (char*)d_ws;
  auto align256 = [](size_t v) { return (v + 255) & ~(size_t)255; };
  size_t off = 0;
  _Float16* m16      = (_Float16*)(ws + off); off = align256(off + (size_t)B_ * T_ * H_ * 2);
  float*    node_max = (float*)(ws + off);    off = align256(off + (size_t)B_ * Ta_ * 4);
  int*      node_idx = (int*)(ws + off);      off = align256(off + (size_t)B_ * Ta_ * 4);
  int*      is_src   = (int*)(ws + off);      off = align256(off + (size_t)B_ * Ta_ * 4);
  int*      pos_unm  = (int*)(ws + off);      off = align256(off + (size_t)B_ * Ta_ * 4);
  int*      pos_mrg  = (int*)(ws + off);      off = align256(off + (size_t)B_ * Ta_ * 4);
  int*      n_unm    = (int*)(ws + off);      off = align256(off + (size_t)B_ * 4);

  tome_normalize_kernel<<<(B_ * T_) / 8, 256, 0, stream>>>(metric, m16);
  tome_match_kernel<<<B_ * (Ta_ / 16), 32, 0, stream>>>(m16, node_max, node_idx);
  tome_rank_kernel<<<(B_ * Ta_) / 256, 256, 0, stream>>>(node_max, rp, is_src);
  tome_scan_kernel<<<B_, 1024, 0, stream>>>(is_src, pos_unm, pos_mrg, n_unm);
  tome_out_even_kernel<<<B_ * Ta_, 256, 0, stream>>>(x, size, is_src, pos_unm,
                                                     pos_mrg, n_unm, out);
  tome_out_dst_kernel<<<B_ * Tb_, 256, 0, stream>>>(x, size, is_src, node_idx,
                                                    n_unm, out);
}
